// FrameRNN_51196010168746
// MI455X (gfx1250) — matrix operand order repaired and checked by reading, not compiled, not run on hardware
//
#include <hip/hip_runtime.h>

// ---------------- problem constants ----------------
#define DIMD     1024
#define B3       3072      // 3*DIMD
#define NB       16        // batch == WMMA M tile
#define TSTEPS   512
#define NLAYERS  4
#define NLOWER   4
#define SEQ      8192
#define TT_OUT   2048      // TSTEPS*NLOWER

// ---------------- workspace layout (bytes) ----------------
#define OFF_WXT   0u                          // [4][3072][1024] bf16  = 25165824
#define OFF_WHT   25165824u                   // [4][3072][1024] bf16  = 25165824
#define OFF_WUPT  50331648u                   // [4][1024][1024] bf16  =  8388608
#define OFF_X0    58720256u                   // [512][16][1024] bf16  = 16777216
#define OFF_SKIP  75497472u                   // [512][16][1024] bf16  = 16777216
#define OFF_HF    92274688u                   // [2][4][16][1024] f32  =   524288
#define OFF_HB    92798976u                   // [2][4][16][1024] bf16 =   262144
#define OFF_BAR   93061120u                   // barrier counters

typedef __attribute__((ext_vector_type(16))) __bf16 v16bf;
typedef __attribute__((ext_vector_type(8)))  __bf16 v8bf;
typedef __attribute__((ext_vector_type(8)))  float  v8f;

__device__ __forceinline__ unsigned short f2bf(float f) {
    unsigned u = __float_as_uint(f);
    u += 0x7FFFu + ((u >> 16) & 1u);          // round-to-nearest-even
    return (unsigned short)(u >> 16);
}

__device__ __forceinline__ v8f wmma_bf16(v16bf a, v16bf b, v8f c) {
    return __builtin_amdgcn_wmma_f32_16x16x32_bf16(false, a, false, b, (short)0, c, false, false);
}

// A operand (16x32 bf16): lane L(0-15): K 0..15 contiguous; lane L+16: K 16..31 contiguous
__device__ __forceinline__ v16bf load_a(const unsigned short* row, int ko, int half) {
    return *(const v16bf*)(row + ko + half * 16);
}
// B operand (32x16 bf16): lane n: K 0..15 contiguous; lane n+16: K 16..31 contiguous
__device__ __forceinline__ v16bf load_b(const unsigned short* row, int ko, int half) {
    return *(const v16bf*)(row + ko + half * 16);
}

__device__ __forceinline__ void grid_barrier(unsigned* cnt, unsigned* gen, unsigned nwg) {
    __threadfence();
    __syncthreads();
    if (threadIdx.x == 0) {
        unsigned g = __hip_atomic_load(gen, __ATOMIC_RELAXED, __HIP_MEMORY_SCOPE_AGENT);
        unsigned arrived = __hip_atomic_fetch_add(cnt, 1u, __ATOMIC_ACQ_REL, __HIP_MEMORY_SCOPE_AGENT);
        if (arrived + 1u == nwg) {
            __hip_atomic_store(cnt, 0u, __ATOMIC_RELAXED, __HIP_MEMORY_SCOPE_AGENT);
            __hip_atomic_fetch_add(gen, 1u, __ATOMIC_RELEASE, __HIP_MEMORY_SCOPE_AGENT);
        } else {
            while (__hip_atomic_load(gen, __ATOMIC_ACQUIRE, __HIP_MEMORY_SCOPE_AGENT) == g)
                __builtin_amdgcn_s_sleep(1);
        }
    }
    __syncthreads();
    __threadfence();
}

// ---------------- prep: weights -> bf16, transposed to [N][K] ----------------
__global__ void wprep_kernel(const float* __restrict__ wx, const float* __restrict__ wh,
                             const float* __restrict__ wup,
                             unsigned short* __restrict__ wxT, unsigned short* __restrict__ whT,
                             unsigned short* __restrict__ wupT) {
    const size_t N1 = (size_t)NLAYERS * DIMD * B3;      // 12582912
    const size_t N3 = (size_t)NLOWER * DIMD * DIMD;     //  4194304
    size_t i = (size_t)blockIdx.x * blockDim.x + threadIdx.x;
    if (i < N1) {                      // wx: [l][k][n] -> wxT[l][n][k]
        size_t l = i / ((size_t)DIMD * B3);
        size_t r = i - l * (size_t)DIMD * B3;
        size_t k = r / B3, n = r - k * B3;
        wxT[l * (size_t)B3 * DIMD + n * DIMD + k] = f2bf(wx[i]);
    } else if (i < 2 * N1) {
        size_t j = i - N1;
        size_t l = j / ((size_t)DIMD * B3);
        size_t r = j - l * (size_t)DIMD * B3;
        size_t k = r / B3, n = r - k * B3;
        whT[l * (size_t)B3 * DIMD + n * DIMD + k] = f2bf(wh[j]);
    } else if (i < 2 * N1 + N3) {      // wup: [k4][i][o] -> wupT[k4][o][i]
        size_t j = i - 2 * N1;
        size_t k4 = j / ((size_t)DIMD * DIMD);
        size_t r  = j - k4 * (size_t)DIMD * DIMD;
        size_t ii = r / DIMD, o = r - ii * DIMD;
        wupT[k4 * (size_t)DIMD * DIMD + o * DIMD + ii] = f2bf(wup[j]);
    }
}

// ---------------- prep: X0[t][b][d] = frames@W_in + b_in + cond (bf16) ----------------
__global__ void x0_kernel(const int* __restrict__ inputs, const float* __restrict__ cond,
                          const float* __restrict__ W_in, const float* __restrict__ b_in,
                          unsigned short* __restrict__ x0) {
    size_t i = (size_t)blockIdx.x * blockDim.x + threadIdx.x;   // over T*NB*D
    int d = (int)(i & (DIMD - 1));
    size_t tb = i >> 10;
    int b = (int)(tb & 15);
    int t = (int)(tb >> 4);
    float acc = b_in[d] + cond[((size_t)b * TSTEPS + t) * DIMD + d];
    const int* fr = inputs + (size_t)b * SEQ + t * 16;
    #pragma unroll
    for (int f = 0; f < 16; ++f) {
        float xv = ((float)fr[f] * (1.0f / 128.0f) - 1.0f) * 2.0f;
        acc += xv * W_in[f * DIMD + d];
    }
    x0[i] = f2bf(acc);
}

// ---------------- prep: zero state + barriers (every call!) ----------------
__global__ void zero_kernel(float* __restrict__ hf, unsigned short* __restrict__ hb,
                            unsigned* __restrict__ bar) {
    int i = blockIdx.x * blockDim.x + threadIdx.x;
    const int NH = 2 * NLAYERS * NB * DIMD;   // 131072
    if (i < NH) { hf[i] = 0.0f; hb[i] = 0; }
    if (i < 64) bar[i] = 0u;
}

// ---------------- persistent GRU-stack recurrence ----------------
// 64 WGs x 64 threads (2 waves). WG w owns output dims [w*16, w*16+16).
// Wave 0: K 0..511, Wave 1: K 512..1023, LDS reduce, wave0 does gate math + h update.
// Ping-pong h buffers -> one grid barrier per layer.
__global__ __launch_bounds__(64) void rnn_persistent(
    const unsigned short* __restrict__ x0,    // [T][16][D]
    const unsigned short* __restrict__ wxT,   // [L][3D][D]
    const unsigned short* __restrict__ whT,   // [L][3D][D]
    const float* __restrict__ bx,             // [L][3D]
    const float* __restrict__ bh,             // [L][3D]
    float* __restrict__ hf,                   // [2][L][16][D] fp32 master
    unsigned short* __restrict__ hb,          // [2][L][16][D] bf16 mirror
    unsigned short* __restrict__ skip,        // [T][16][D]
    unsigned* __restrict__ bar, int nwg) {
    __shared__ float red[4 * 8 * 32];

    const int tid  = threadIdx.x;
    const int wave = tid >> 5;
    const int lane = tid & 31;
    const int half = lane >> 4;
    const int idx  = lane & 15;
    const int d    = blockIdx.x * 16 + idx;   // output dim owned by this lane
    const int kb0  = wave * 16;               // 16 k-iters of 32 per wave

    for (int t = 0; t < TSTEPS; ++t) {
        const int cur = t & 1, prv = cur ^ 1;
        float skipacc[8] = {0.f,0.f,0.f,0.f,0.f,0.f,0.f,0.f};
        const unsigned short* inp = x0 + (size_t)t * NB * DIMD;

        for (int l = 0; l < NLAYERS; ++l) {
            const size_t lw = (size_t)l * B3 * DIMD;
            const unsigned short* wxz = wxT + lw + (size_t)(0 * DIMD + d) * DIMD;
            const unsigned short* wxr = wxT + lw + (size_t)(1 * DIMD + d) * DIMD;
            const unsigned short* wxh = wxT + lw + (size_t)(2 * DIMD + d) * DIMD;
            const unsigned short* whz = whT + lw + (size_t)(0 * DIMD + d) * DIMD;
            const unsigned short* whr = whT + lw + (size_t)(1 * DIMD + d) * DIMD;
            const unsigned short* whh = whT + lw + (size_t)(2 * DIMD + d) * DIMD;
            const unsigned short* arow = inp + (size_t)idx * DIMD;
            const unsigned short* hrow = hb + ((size_t)(prv * NLAYERS + l) * NB + idx) * DIMD;

            v8f az  = {0.f,0.f,0.f,0.f,0.f,0.f,0.f,0.f};
            v8f ar_ = {0.f,0.f,0.f,0.f,0.f,0.f,0.f,0.f};
            v8f axh = {0.f,0.f,0.f,0.f,0.f,0.f,0.f,0.f};
            v8f ahh = {0.f,0.f,0.f,0.f,0.f,0.f,0.f,0.f};

            #pragma unroll 4
            for (int kb = kb0; kb < kb0 + 16; ++kb) {
                const int ko = kb * 32;
                v16bf ax = load_a(arow, ko, half);
                v16bf ah = load_a(hrow, ko, half);
                az  = wmma_bf16(ax, load_b(wxz, ko, half), az);
                az  = wmma_bf16(ah, load_b(whz, ko, half), az);
                ar_ = wmma_bf16(ax, load_b(wxr, ko, half), ar_);
                ar_ = wmma_bf16(ah, load_b(whr, ko, half), ar_);
                axh = wmma_bf16(ax, load_b(wxh, ko, half), axh);
                ahh = wmma_bf16(ah, load_b(whh, ko, half), ahh);
            }

            if (wave == 1) {
                #pragma unroll
                for (int i = 0; i < 8; ++i) {
                    red[(0 * 8 + i) * 32 + lane] = az[i];
                    red[(1 * 8 + i) * 32 + lane] = ar_[i];
                    red[(2 * 8 + i) * 32 + lane] = axh[i];
                    red[(3 * 8 + i) * 32 + lane] = ahh[i];
                }
            }
            __syncthreads();

            if (wave == 0) {
                const size_t l3 = (size_t)l * B3;
                const float bzv  = bx[l3 + d] + bh[l3 + d];
                const float brv  = bx[l3 + DIMD + d] + bh[l3 + DIMD + d];
                const float bxhv = bx[l3 + 2 * DIMD + d];
                const float bhhv = bh[l3 + 2 * DIMD + d];
                const float* hof = hf + (size_t)(prv * NLAYERS + l) * NB * DIMD;
                float* hnf = hf + (size_t)(cur * NLAYERS + l) * NB * DIMD;
                unsigned short* hnb = hb + (size_t)(cur * NLAYERS + l) * NB * DIMD;
                #pragma unroll
                for (int i = 0; i < 8; ++i) {
                    const int m = i + 8 * half;              // batch row (C layout)
                    float zz = az[i]  + red[(0 * 8 + i) * 32 + lane] + bzv;
                    float rr = ar_[i] + red[(1 * 8 + i) * 32 + lane] + brv;
                    float xh = axh[i] + red[(2 * 8 + i) * 32 + lane] + bxhv;
                    float hh = ahh[i] + red[(3 * 8 + i) * 32 + lane] + bhhv;
                    float zg = 1.0f / (1.0f + __expf(-zz));
                    float rg = 1.0f / (1.0f + __expf(-rr));
                    float cand = tanhf(xh + rg * hh);
                    size_t off = (size_t)m * DIMD + d;
                    float ho = hof[off];
                    float hn = zg * ho + (1.0f - zg) * cand;
                    hnf[off] = hn;
                    hnb[off] = f2bf(hn);
                    skipacc[i] += hn;
                }
            }
            grid_barrier(bar, bar + 1, (unsigned)nwg);
            inp = hb + (size_t)(cur * NLAYERS + l) * NB * DIMD;   // next layer input = h_l(t)
        }

        if (wave == 0) {
            unsigned short* sp = skip + (size_t)t * NB * DIMD;
            #pragma unroll
            for (int i = 0; i < 8; ++i)
                sp[(size_t)(i + 8 * half) * DIMD + d] = f2bf(skipacc[i]);
        }
    }
}

// ---------------- ConvT upsample: out[b][t*4+k][o] = skip[t][b]@W_up[k] + b_up ----------------
__global__ __launch_bounds__(128) void upsample_kernel(
    const unsigned short* __restrict__ skip,   // [T][16][D] bf16
    const unsigned short* __restrict__ wupT,   // [4][D_out][D_in] bf16
    const float* __restrict__ bup,
    float* __restrict__ out) {                 // [16][2048][1024] f32
    const int wave = threadIdx.x >> 5;
    const int lane = threadIdx.x & 31;
    const int half = lane >> 4, idx = lane & 15;
    const int tile = blockIdx.x * 4 + wave;    // 512*4*64 tiles
    const int t  = tile >> 8;
    const int rm = tile & 255;
    const int k  = rm >> 6;
    const int oc = rm & 63;
    const int o  = oc * 16 + idx;

    const unsigned short* arow = skip + ((size_t)t * NB + idx) * DIMD;
    const unsigned short* brow = wupT + ((size_t)k * DIMD + o) * DIMD;
    v8f acc = {0.f,0.f,0.f,0.f,0.f,0.f,0.f,0.f};
    #pragma unroll 8
    for (int kb = 0; kb < 32; ++kb) {
        const int ko = kb * 32;
        acc = wmma_bf16(load_a(arow, ko, half), load_b(brow, ko, half), acc);
    }
    const float bv = bup[o];
    const int tt = t * 4 + k;
    #pragma unroll
    for (int i = 0; i < 8; ++i) {
        const int m = i + 8 * half;
        out[((size_t)m * TT_OUT + tt) * DIMD + o] = acc[i] + bv;
    }
}

extern "C" void kernel_launch(void* const* d_in, const int* in_sizes, int n_in,
                              void* d_out, int out_size, void* d_ws, size_t ws_size,
                              hipStream_t stream) {
    (void)in_sizes; (void)n_in; (void)out_size; (void)ws_size;
    const int*   inputs = (const int*)  d_in[0];
    const float* cond   = (const float*)d_in[1];
    const float* W_in   = (const float*)d_in[2];
    const float* b_in   = (const float*)d_in[3];
    const float* wx     = (const float*)d_in[4];
    const float* wh     = (const float*)d_in[5];
    const float* bx     = (const float*)d_in[6];
    const float* bh     = (const float*)d_in[7];
    const float* wup    = (const float*)d_in[8];
    const float* bup    = (const float*)d_in[9];
    float* out = (float*)d_out;

    char* ws = (char*)d_ws;
    unsigned short* wxT  = (unsigned short*)(ws + OFF_WXT);
    unsigned short* whT  = (unsigned short*)(ws + OFF_WHT);
    unsigned short* wupT = (unsigned short*)(ws + OFF_WUPT);
    unsigned short* x0   = (unsigned short*)(ws + OFF_X0);
    unsigned short* skip = (unsigned short*)(ws + OFF_SKIP);
    float*          hf   = (float*)(ws + OFF_HF);
    unsigned short* hb   = (unsigned short*)(ws + OFF_HB);
    unsigned*       bar  = (unsigned*)(ws + OFF_BAR);

    wprep_kernel<<<114688, 256, 0, stream>>>(wx, wh, wup, wxT, whT, wupT);
    x0_kernel<<<32768, 256, 0, stream>>>(inputs, cond, W_in, b_in, x0);
    zero_kernel<<<512, 256, 0, stream>>>(hf, hb, bar);
    rnn_persistent<<<64, 64, 0, stream>>>(x0, wxT, whT, bx, bh, hf, hb, skip, bar, 64);
    upsample_kernel<<<32768, 128, 0, stream>>>(skip, wupT, bup, out);
}
